// AIMPretrainer_32126355374833
// MI455X (gfx1250) — matrix operand order repaired and checked by reading, not compiled, not run on hardware
//
#include <hip/hip_runtime.h>

typedef __attribute__((ext_vector_type(16))) __bf16 bf16x16;
typedef __attribute__((ext_vector_type(8)))  __bf16 bf16x8;
typedef __attribute__((ext_vector_type(8)))  float  f32x8;

namespace {
constexpr int kB   = 4;
constexpr int kL   = 32768;
constexpr int kN   = 2048;    // patches
constexpr int kNT  = 2049;    // tokens (+CLS)
constexpr int kNTP = 2112;    // padded tokens, multiple of 64
constexpr int kD   = 384;
constexpr int kH   = 6;
constexpr int kFF  = 1536;
constexpr float kMISS = -999.0f;

constexpr size_t OUT_REC  = 1;
constexpr size_t OUT_ART  = OUT_REC + (size_t)kB * kN * 16;   // 131073
constexpr size_t OUT_INH  = OUT_ART + (size_t)kB * kN;        // 139265
constexpr size_t OUT_DROP = OUT_INH + (size_t)kB * kNT;       // 147461
}

// ---------------------------------------------------------------- utilities
__global__ void k_f2b(const float* __restrict__ s, __bf16* __restrict__ d, int n) {
  int i = blockIdx.x * 256 + threadIdx.x;
  if (i < n) d[i] = (__bf16)s[i];
}

__global__ void k_zero1(float* a) { a[0] = 0.0f; }

// ------------------------------------------------------------ mask pipeline
__global__ void k_patchstats(const float* __restrict__ x, float* __restrict__ mean,
                             float* __restrict__ rvt, unsigned char* __restrict__ inh) {
  int t = blockIdx.x * 256 + threadIdx.x;
  if (t >= kB * kN) return;
  int b = t / kN, n = t % kN;
  const float* xp = x + (size_t)b * kL + (size_t)n * 16;
  float v[16]; float s = 0.f; int miss = 0;
  #pragma unroll
  for (int p = 0; p < 16; ++p) { v[p] = xp[p]; s += v[p]; if (v[p] == kMISS) miss++; }
  float m = s * (1.f / 16.f), ss = 0.f;
  #pragma unroll
  for (int p = 0; p < 16; ++p) { float d = v[p] - m; ss += d * d; }
  mean[t] = m;
  rvt[t]  = rsqrtf(ss * (1.f / 15.f) + 1e-6f);   // unbiased var
  inh[t]  = (miss >= 8) ? 1 : 0;
}

__global__ void k_counts(const unsigned char* __restrict__ inh, int* __restrict__ numart) {
  int b = threadIdx.x;
  if (b >= kB) return;
  int c = 0;
  for (int n = 0; n < kN; ++n) c += inh[b * kN + n];
  float na = fmaxf(0.f, 0.5f - (float)c / (float)kN) * (float)kN;
  int v = (int)na;               // trunc, like astype(int32)
  int cap = kN - c;
  numart[b] = v < cap ? v : cap;
}

__global__ void k_rank_art(const float* __restrict__ sc, const unsigned char* __restrict__ inh,
                           const int* __restrict__ numart, unsigned char* __restrict__ art) {
  __shared__ float s[kN];
  int b = blockIdx.x;
  for (int i = threadIdx.x; i < kN; i += 256)
    s[i] = inh[b * kN + i] ? __builtin_inff() : sc[b * kN + i];
  __syncthreads();
  int na = numart[b];
  for (int i = threadIdx.x; i < kN; i += 256) {
    float si = s[i]; int r = 0;
    for (int j = 0; j < kN; ++j) {
      float sj = s[j];
      r += (sj < si) || (sj == si && j < i);   // stable rank (argsort∘argsort)
    }
    art[b * kN + i] = (r < na) ? 1 : 0;
  }
}

__global__ void k_rank_drop(const float* __restrict__ ds, const unsigned char* __restrict__ inh,
                            const unsigned char* __restrict__ art, const int* __restrict__ numart,
                            unsigned char* __restrict__ comb, unsigned char* __restrict__ drop) {
  __shared__ float s[kNT];
  int b = blockIdx.x;
  for (int t = threadIdx.x; t < kNT; t += 256) {
    unsigned char cb = (t == 0) ? 0 : (unsigned char)(inh[b * kN + t - 1] | art[b * kN + t - 1]);
    comb[b * kNT + t] = cb;
    s[t] = cb ? ds[b * kNT + t] : __builtin_inff();
  }
  __syncthreads();
  int na = numart[b];
  for (int t = threadIdx.x; t < kNT; t += 256) {
    float si = s[t]; int r = 0;
    for (int j = 0; j < kNT; ++j) {
      float sj = s[j];
      r += (sj < si) || (sj == si && j < t);
    }
    drop[b * kNT + t] = (r < na) ? 1 : 0;
  }
}

__global__ void k_order(const unsigned char* __restrict__ drop, int* __restrict__ order,
                        int* __restrict__ invp) {
  __shared__ unsigned char df[kNT];
  __shared__ int nk;
  int b = blockIdx.x;
  for (int t = threadIdx.x; t < kNT; t += 256) df[t] = drop[b * kNT + t];
  __syncthreads();
  if (threadIdx.x == 0) {
    int c = 0;
    for (int t = 0; t < kNT; ++t) c += df[t];
    nk = kNT - c;
  }
  __syncthreads();
  for (int t = threadIdx.x; t < kNT; t += 256) {
    int cd = 0;
    for (int j = 0; j < t; ++j) cd += df[j];
    int pos = df[t] ? (nk + cd) : (t - cd);    // stable kept-first partition
    order[b * kNTP + pos] = t;
    invp[b * kNTP + t]    = pos;
  }
}

// ----------------------------------------------------------- embed / gather
__global__ void k_embed(const float* __restrict__ x, const float* __restrict__ We,
                        const float* __restrict__ be, const float* __restrict__ cls,
                        const float* __restrict__ pos, float* __restrict__ emb) {
  size_t i = (size_t)blockIdx.x * 256 + threadIdx.x;
  size_t tot = (size_t)kB * kNTP * kD;
  if (i >= tot) return;
  int d = (int)(i % kD);
  size_t bt = i / kD;
  int t = (int)(bt % kNTP), b = (int)(bt / kNTP);
  float v;
  if (t >= kNT) v = 0.f;
  else if (t == 0) v = cls[d] + pos[d];
  else {
    const float* xp = x + (size_t)b * kL + (size_t)(t - 1) * 16;
    float s = be[d];
    #pragma unroll
    for (int p = 0; p < 16; ++p) s += xp[p] * We[p * kD + d];
    v = s + pos[(size_t)t * kD + d];
  }
  emb[i] = v;
}

__global__ void k_gather(const float* __restrict__ emb, const int* __restrict__ order,
                         const unsigned char* __restrict__ comb, const unsigned char* __restrict__ drop,
                         const float* __restrict__ mtok, float* __restrict__ xk,
                         unsigned char* __restrict__ amask) {
  size_t i = (size_t)blockIdx.x * 256 + threadIdx.x;
  size_t tot = (size_t)kB * kNTP * kD;
  if (i >= tot) return;
  int d = (int)(i % kD);
  size_t bt = i / kD;
  int j = (int)(bt % kNTP), b = (int)(bt / kNTP);
  float v = 0.f; unsigned char am = 1;
  if (j < kNT) {
    int t = order[b * kNTP + j];
    unsigned char dr = drop[b * kNT + t], cb = comb[b * kNT + t];
    if (!dr) { am = cb; v = cb ? mtok[d] : emb[((size_t)b * kNTP + t) * kD + d]; }
  }
  xk[i] = v;
  if (d == 0) amask[b * kNTP + j] = am;
}

__global__ void k_reinsert(const float* __restrict__ emb, const float* __restrict__ hs,
                           const int* __restrict__ invp, const unsigned char* __restrict__ drop,
                           float* __restrict__ full) {
  size_t i = (size_t)blockIdx.x * 256 + threadIdx.x;
  size_t tot = (size_t)kB * kNTP * kD;
  if (i >= tot) return;
  int d = (int)(i % kD);
  size_t bt = i / kD;
  int t = (int)(bt % kNTP), b = (int)(bt / kNTP);
  float v = 0.f;
  if (t < kNT) {
    if (drop[b * kNT + t]) v = emb[i];
    else v = hs[((size_t)b * kNTP + invp[b * kNTP + t]) * kD + d];
  }
  full[i] = v;
}

// ------------------------------------------------------------------ LayerNorm
__global__ void k_ln(const float* __restrict__ X, const float* __restrict__ g,
                     const float* __restrict__ bb, __bf16* __restrict__ Y, int rows) {
  int wave = threadIdx.x >> 5, lane = threadIdx.x & 31;
  int row = blockIdx.x * 8 + wave;
  if (row >= rows) return;
  const float* xr = X + (size_t)row * kD;
  float v[12], s = 0.f, ss = 0.f;
  #pragma unroll
  for (int e = 0; e < 12; ++e) { v[e] = xr[lane + 32 * e]; s += v[e]; ss += v[e] * v[e]; }
  #pragma unroll
  for (int m = 16; m; m >>= 1) { s += __shfl_xor(s, m, 32); ss += __shfl_xor(ss, m, 32); }
  float mean = s * (1.f / kD);
  float var  = ss * (1.f / kD) - mean * mean;
  float rs = rsqrtf(var + 1e-5f);
  __bf16* yr = Y + (size_t)row * kD;
  #pragma unroll
  for (int e = 0; e < 12; ++e) {
    int d = lane + 32 * e;
    yr[d] = (__bf16)((v[e] - mean) * rs * g[d] + bb[d]);
  }
}

// ------------------------------------------------------- generic bf16 WMMA GEMM
// C[M,N] = A[M,K](bf16) @ W[K,N](bf16 row-major) [+bias] [gelu] [+resid] -> f32 or bf16
// block = 128 threads = 4 waves along N; wave computes a 32 x (16*SUB) tile
// (2 x SUB WMMA grid, branch-free inner loop). grid = (ceil(N/(64*SUB)), M/32).
// Requires: M % 32 == 0 and N % (16*SUB) == 0.
template <int SUB>
__global__ void k_gemm(const __bf16* __restrict__ A, int lda,
                       const __bf16* __restrict__ W, int ldw,
                       const float* __restrict__ bias,
                       const float* __restrict__ resid,
                       float* __restrict__ Cf, __bf16* __restrict__ Cb, int ldc,
                       int M, int N, int K, int act) {
  int lane = threadIdx.x & 31;
  int wave = threadIdx.x >> 5;
  int n0 = (blockIdx.x * 4 + wave) * (16 * SUB);
  int m0 = blockIdx.y * 32;
  if (n0 >= N) return;                      // wave-uniform

  // A frag: lanes 0-15 rows m, K=0..7 & 16..23; lanes 16-31 same rows, K=8..15 & 24..31
  int akb = (lane >> 4) * 8;
  const __bf16* Ar0 = A + (size_t)(m0 + (lane & 15)) * lda + akb;
  const __bf16* Ar1 = Ar0 + (size_t)16 * lda;
  // B frag: lane%16 = N column; lanes 0-15 K=0..7 & 16..23; lanes 16-31 K=8..15 & 24..31
  const __bf16* Wl = W + (size_t)akb * ldw + n0 + (lane & 15);

  f32x8 acc[2][SUB];
  #pragma unroll
  for (int r = 0; r < 2; ++r)
    #pragma unroll
    for (int s = 0; s < SUB; ++s)
      acc[r][s] = (f32x8){0.f, 0.f, 0.f, 0.f, 0.f, 0.f, 0.f, 0.f};

  for (int k0 = 0; k0 < K; k0 += 32) {
    __builtin_prefetch(Wl + (size_t)(k0 + 32) * ldw, 0, 1);
    bf16x8 a0lo = *(const bf16x8*)(Ar0 + k0);
    bf16x8 a0hi = *(const bf16x8*)(Ar0 + k0 + 16);
    bf16x8 a1lo = *(const bf16x8*)(Ar1 + k0);
    bf16x8 a1hi = *(const bf16x8*)(Ar1 + k0 + 16);
    bf16x16 a0, a1;
    #pragma unroll
    for (int i = 0; i < 8; ++i) {
      a0[i] = a0lo[i]; a0[i + 8] = a0hi[i];
      a1[i] = a1lo[i]; a1[i + 8] = a1hi[i];
    }
    const __bf16* Wk = Wl + (size_t)k0 * ldw;
    #pragma unroll
    for (int s = 0; s < SUB; ++s) {
      bf16x16 bfr;
      #pragma unroll
      for (int i = 0; i < 8; ++i) {
        bfr[i] = Wk[(size_t)i * ldw + 16 * s];
        bfr[i + 8] = Wk[(size_t)(16 + i) * ldw + 16 * s];
      }
      acc[0][s] = __builtin_amdgcn_wmma_f32_16x16x32_bf16(false, a0, false, bfr,
                                                          (short)0, acc[0][s], false, false);
      acc[1][s] = __builtin_amdgcn_wmma_f32_16x16x32_bf16(false, a1, false, bfr,
                                                          (short)0, acc[1][s], false, false);
    }
  }

  int ch = lane & 15;
  int rh = (lane >> 4) << 3;
  #pragma unroll
  for (int s = 0; s < SUB; ++s) {
    int col = n0 + 16 * s + ch;
    float bv = bias ? bias[col] : 0.f;
    #pragma unroll
    for (int r = 0; r < 2; ++r) {
      #pragma unroll
      for (int i = 0; i < 8; ++i) {
        int row = m0 + 16 * r + rh + i;
        float v = acc[r][s][i] + bv;
        if (act == 1) {  // tanh-approx GELU (jax.nn.gelu default)
          float c = v + 0.044715f * v * v * v;
          v = 0.5f * v * (1.f + tanhf(0.7978845608028654f * c));
        }
        size_t off = (size_t)row * ldc + col;
        if (resid) v += resid[off];
        if (Cf) Cf[off] = v; else Cb[off] = (__bf16)v;
      }
    }
  }
}

// ------------------------------------------------------------------ attention
__global__ void k_split(const float* __restrict__ qkv, __bf16* __restrict__ Qb,
                        __bf16* __restrict__ Kt, __bf16* __restrict__ Vb) {
  size_t i = (size_t)blockIdx.x * 256 + threadIdx.x;
  size_t tot = (size_t)kB * kNTP * kD;
  if (i >= tot) return;
  int di = (int)(i % kD);
  size_t bt = i / kD;
  int t = (int)(bt % kNTP), b = (int)(bt / kNTP);
  int h = di >> 6, ii = di & 63;
  const float* base = qkv + ((size_t)b * kNTP + t) * (3 * kD);
  size_t bh = (size_t)(b * kH + h);
  Qb[(bh * kNTP + t) * 64 + ii]  = (__bf16)(base[di] * 0.125f);   // / sqrt(64)
  Kt[(bh * 64 + ii) * kNTP + t]  = (__bf16)(base[kD + di]);       // transposed
  Vb[(bh * kNTP + t) * 64 + ii]  = (__bf16)(base[2 * kD + di]);
}

__global__ void k_softmax(const float* __restrict__ S, __bf16* __restrict__ Pb,
                          const unsigned char* __restrict__ mask /* nullable, len kNTP */) {
  __shared__ float row[kNTP];
  __shared__ float red[256];
  int r = blockIdx.x, tid = threadIdx.x;
  const float* Sr = S + (size_t)r * kNTP;
  for (int j = tid; j < kNTP; j += 256) {
    bool bad = (j >= kNT) || (mask && mask[j]);
    row[j] = bad ? -3.0e38f : Sr[j];
  }
  __syncthreads();
  float mx = -3.0e38f;
  for (int j = tid; j < kNTP; j += 256) mx = fmaxf(mx, row[j]);
  red[tid] = mx; __syncthreads();
  for (int s = 128; s > 0; s >>= 1) { if (tid < s) red[tid] = fmaxf(red[tid], red[tid + s]); __syncthreads(); }
  mx = red[0]; __syncthreads();
  float sm = 0.f;
  for (int j = tid; j < kNTP; j += 256) {
    float e = (row[j] <= -1.0e38f) ? 0.f : expf(row[j] - mx);
    row[j] = e; sm += e;
  }
  red[tid] = sm; __syncthreads();
  for (int s = 128; s > 0; s >>= 1) { if (tid < s) red[tid] += red[tid + s]; __syncthreads(); }
  float inv = 1.f / red[0];
  __bf16* Pr = Pb + (size_t)r * kNTP;
  for (int j = tid; j < kNTP; j += 256) Pr[j] = (__bf16)(row[j] * inv);
}

// ------------------------------------------------------------------- outputs
__global__ void k_out_rec(const float* __restrict__ rec, const unsigned char* __restrict__ art,
                          const float* __restrict__ x, const float* __restrict__ mean,
                          const float* __restrict__ rvt, float* __restrict__ out,
                          float* __restrict__ acc) {
  int t = blockIdx.x * 256 + threadIdx.x;
  if (t >= kB * kN) return;
  int b = t / kN, n = t % kN;
  const float* rp = rec + ((size_t)b * kNTP + (n + 1)) * 16;
  float* op = out + OUT_REC + (size_t)t * 16;
  unsigned char a = art[t];
  float m = mean[t], rv = rvt[t];
  const float* xp = x + (size_t)b * kL + (size_t)n * 16;
  float s = 0.f;
  #pragma unroll
  for (int p = 0; p < 16; ++p) {
    float r = rp[p];
    op[p] = r;
    float tg = (xp[p] - m) * rv;
    float df = r - tg;
    s += df * df;
  }
  out[OUT_ART + t] = a ? 1.f : 0.f;
  if (a) atomicAdd(acc, s);
}

__global__ void k_out_misc(const unsigned char* __restrict__ inh,
                           const unsigned char* __restrict__ drop, float* __restrict__ out) {
  int t = blockIdx.x * 256 + threadIdx.x;
  if (t >= kB * kNT) return;
  int b = t / kNT, tt = t % kNT;
  out[OUT_INH + t]  = (tt == 0) ? 0.f : (inh[b * kN + tt - 1] ? 1.f : 0.f);
  out[OUT_DROP + t] = drop[t] ? 1.f : 0.f;
}

__global__ void k_loss_fin(const float* __restrict__ acc, const int* __restrict__ numart,
                           float* __restrict__ out) {
  int c = numart[0] + numart[1] + numart[2] + numart[3];
  out[0] = acc[0] / fmaxf((float)c * 16.f, 1.f);
}

// ------------------------------------------------------------------- launch
extern "C" void kernel_launch(void* const* d_in, const int* in_sizes, int n_in,
                              void* d_out, int out_size, void* d_ws, size_t ws_size,
                              hipStream_t stream) {
  (void)in_sizes; (void)n_in; (void)out_size; (void)ws_size;
  const float* x   = (const float*)d_in[0];
  const float* as  = (const float*)d_in[1];
  const float* dsc = (const float*)d_in[2];
  const float* We  = (const float*)d_in[3];
  const float* be  = (const float*)d_in[4];
  const float* cls = (const float*)d_in[5];
  const float* pos = (const float*)d_in[6];
  const float* mtok= (const float*)d_in[7];
  const float* Wh  = (const float*)d_in[8];
  const float* bh  = (const float*)d_in[9];
  // blocks (insertion order): ln1g,ln1b,Wqkv,bqkv,Wo,bo,ln2g,ln2b,W1,b1,W2,b2
  const int lb[3] = {10, 22, 34};            // enc0, enc1, dec0
  float* out = (float*)d_out;

  // ---- workspace bump allocator
  char* wp = (char*)d_ws;
  auto alloc = [&](size_t bytes) -> void* {
    void* r = (void*)wp;
    wp += (bytes + 255) & ~(size_t)255;
    return r;
  };
  const size_t TOKD = (size_t)kB * kNTP * kD;
  float*  emb  = (float*) alloc(TOKD * 4);
  float*  strm = (float*) alloc(TOKD * 4);
  float*  full = (float*) alloc(TOKD * 4);
  float*  qkv  = (float*) alloc((size_t)kB * kNTP * 3 * kD * 4);
  __bf16* lnb  = (__bf16*)alloc(TOKD * 2);
  __bf16* ffb  = (__bf16*)alloc((size_t)kB * kNTP * kFF * 2);
  __bf16* attO = (__bf16*)alloc(TOKD * 2);
  __bf16* Qb   = (__bf16*)alloc((size_t)kB * kH * kNTP * 64 * 2);
  __bf16* Kt   = (__bf16*)alloc((size_t)kB * kH * kNTP * 64 * 2);
  __bf16* Vb   = (__bf16*)alloc((size_t)kB * kH * kNTP * 64 * 2);
  float*  Sb   = (float*) alloc((size_t)kNTP * kNTP * 4);
  __bf16* Pb   = (__bf16*)alloc((size_t)kNTP * kNTP * 2);
  float*  rec  = (float*) alloc((size_t)kB * kNTP * 16 * 4);
  __bf16* wqB[3]; __bf16* woB[3]; __bf16* w1B[3]; __bf16* w2B[3];
  for (int l = 0; l < 3; ++l) {
    wqB[l] = (__bf16*)alloc((size_t)kD * 3 * kD * 2);
    woB[l] = (__bf16*)alloc((size_t)kD * kD * 2);
    w1B[l] = (__bf16*)alloc((size_t)kD * kFF * 2);
    w2B[l] = (__bf16*)alloc((size_t)kFF * kD * 2);
  }
  __bf16* whB  = (__bf16*)alloc((size_t)kD * 16 * 2);
  float*  mean = (float*) alloc((size_t)kB * kN * 4);
  float*  rvt  = (float*) alloc((size_t)kB * kN * 4);
  unsigned char* inh   = (unsigned char*)alloc(kB * kN);
  unsigned char* art   = (unsigned char*)alloc(kB * kN);
  unsigned char* comb  = (unsigned char*)alloc(kB * kNT);
  unsigned char* drop  = (unsigned char*)alloc(kB * kNT);
  unsigned char* amask = (unsigned char*)alloc(kB * kNTP);
  int*   order = (int*)  alloc((size_t)kB * kNTP * 4);
  int*   invp  = (int*)  alloc((size_t)kB * kNTP * 4);
  int*   numart= (int*)  alloc(kB * 4);
  float* acc   = (float*)alloc(4);

  auto cvt = [&](const float* s, __bf16* d, int n) {
    k_f2b<<<(n + 255) / 256, 256, 0, stream>>>(s, d, n);
  };
  for (int l = 0; l < 3; ++l) {
    cvt((const float*)d_in[lb[l] + 2],  wqB[l], kD * 3 * kD);
    cvt((const float*)d_in[lb[l] + 4],  woB[l], kD * kD);
    cvt((const float*)d_in[lb[l] + 8],  w1B[l], kD * kFF);
    cvt((const float*)d_in[lb[l] + 10], w2B[l], kFF * kD);
  }
  cvt(Wh, whB, kD * 16);

  // ---- masks / ordering / embedding
  k_patchstats<<<(kB * kN + 255) / 256, 256, 0, stream>>>(x, mean, rvt, inh);
  k_counts<<<1, 32, 0, stream>>>(inh, numart);
  k_rank_art<<<kB, 256, 0, stream>>>(as, inh, numart, art);
  k_rank_drop<<<kB, 256, 0, stream>>>(dsc, inh, art, numart, comb, drop);
  k_order<<<kB, 256, 0, stream>>>(drop, order, invp);
  int nTokD = (int)((TOKD + 255) / 256);
  k_embed<<<nTokD, 256, 0, stream>>>(x, We, be, cls, pos, emb);
  k_gather<<<nTokD, 256, 0, stream>>>(emb, order, comb, drop, mtok, strm, amask);

  const int M = kB * kNTP;
  // SUB=4 requires N%64==0 (used for N=1152,384,1536,2112); SUB=1 for narrow N (64,16)
  auto gemm = [&](const __bf16* A, int lda, const __bf16* W, int ldw,
                  const float* bias, const float* resid,
                  float* Cf, __bf16* Cb, int ldc, int Mm, int Nn, int Kk, int actf) {
    if (Nn >= 128) {
      dim3 g((Nn + 255) / 256, Mm / 32);
      k_gemm<4><<<g, 128, 0, stream>>>(A, lda, W, ldw, bias, resid, Cf, Cb, ldc, Mm, Nn, Kk, actf);
    } else {
      dim3 g((Nn + 63) / 64, Mm / 32);
      k_gemm<1><<<g, 128, 0, stream>>>(A, lda, W, ldw, bias, resid, Cf, Cb, ldc, Mm, Nn, Kk, actf);
    }
  };

  auto run_block = [&](float* sbuf, int l, bool use_mask) {
    const float* ln1g = (const float*)d_in[lb[l] + 0];
    const float* ln1b = (const float*)d_in[lb[l] + 1];
    const float* bqkv = (const float*)d_in[lb[l] + 3];
    const float* bo   = (const float*)d_in[lb[l] + 5];
    const float* ln2g = (const float*)d_in[lb[l] + 6];
    const float* ln2b = (const float*)d_in[lb[l] + 7];
    const float* b1   = (const float*)d_in[lb[l] + 9];
    const float* b2   = (const float*)d_in[lb[l] + 11];

    k_ln<<<(M + 7) / 8, 256, 0, stream>>>(sbuf, ln1g, ln1b, lnb, M);
    gemm(lnb, kD, wqB[l], 3 * kD, bqkv, nullptr, qkv, nullptr, 3 * kD, M, 3 * kD, kD, 0);
    k_split<<<nTokD, 256, 0, stream>>>(qkv, Qb, Kt, Vb);
    for (int bhh = 0; bhh < kB * kH; ++bhh) {
      int b = bhh / kH, h = bhh % kH;
      size_t off = (size_t)bhh * kNTP * 64;
      gemm(Qb + off, 64, Kt + off, kNTP, nullptr, nullptr, Sb, nullptr, kNTP,
           kNTP, kNTP, 64, 0);
      k_softmax<<<kNTP, 256, 0, stream>>>(Sb, Pb, use_mask ? amask + b * kNTP : nullptr);
      gemm(Pb, kNTP, Vb + off, 64, nullptr, nullptr, nullptr,
           attO + (size_t)b * kNTP * kD + h * 64, kD, kNTP, 64, kNTP, 0);
    }
    gemm(attO, kD, woB[l], kD, bo, sbuf, sbuf, nullptr, kD, M, kD, kD, 0);
    k_ln<<<(M + 7) / 8, 256, 0, stream>>>(sbuf, ln2g, ln2b, lnb, M);
    gemm(lnb, kD, w1B[l], kFF, b1, nullptr, nullptr, ffb, kFF, M, kFF, kD, 1);
    gemm(ffb, kFF, w2B[l], kD, b2, sbuf, sbuf, nullptr, kD, M, kD, kFF, 0);
  };

  run_block(strm, 0, true);    // enc layer 0
  run_block(strm, 1, true);    // enc layer 1
  k_reinsert<<<nTokD, 256, 0, stream>>>(emb, strm, invp, drop, full);
  run_block(full, 2, false);   // dec layer 0

  // head: rec = d @ Wh + bh
  k_f2b<<<nTokD, 256, 0, stream>>>(full, lnb, (int)TOKD);
  gemm(lnb, kD, whB, 16, bh, nullptr, rec, nullptr, 16, M, 16, kD, 0);

  // outputs + loss
  k_zero1<<<1, 1, 0, stream>>>(acc);
  k_out_rec<<<(kB * kN + 255) / 256, 256, 0, stream>>>(rec, art, x, mean, rvt, out, acc);
  k_out_misc<<<(kB * kNT + 255) / 256, 256, 0, stream>>>(inh, drop, out);
  k_loss_fin<<<1, 1, 0, stream>>>(acc, numart, out);
}
